// SelfAttention_67267777790364
// MI455X (gfx1250) — compile-verified, matrix operand order checked by reading, not run on hardware
//
#include <hip/hip_runtime.h>
#include <hip/hip_bf16.h>
#include <stddef.h>

// ---------------------------------------------------------------------------
// Types for CDNA5 WMMA (wave32): A/B fragments are v16h (8 VGPRs), C/D v8f.
// ---------------------------------------------------------------------------
typedef _Float16 h4   __attribute__((ext_vector_type(4)));
typedef _Float16 h8   __attribute__((ext_vector_type(8)));
typedef _Float16 v16h __attribute__((ext_vector_type(16)));
typedef float    v8f  __attribute__((ext_vector_type(8)));

#define S_LEN 2048
#define HID   4096
#define NQH   32
#define NKVG  8
#define HD    128
#define OQKV  6144   // (32 + 2*8) * 128

__device__ __forceinline__ v8f vzero8() {
  v8f z;
#pragma unroll
  for (int e = 0; e < 8; ++e) z[e] = 0.0f;
  return z;
}

// Build a 16x32 f16 A/B fragment for one lane: per CDNA5 ISA layout the lane
// holds two contiguous 8-halfword (16B) chunks of its row, 16 K apart.
__device__ __forceinline__ v16h frag_ld(const _Float16* p0, const _Float16* p1) {
  h8 lo = *(const h8*)p0;
  h8 hi = *(const h8*)p1;
  return __builtin_shufflevector(lo, hi, 0,1,2,3,4,5,6,7,8,9,10,11,12,13,14,15);
}

__device__ __forceinline__ v8f wmma16(v16h a, v16h b, v8f c) {
  // (neg_a, A, neg_b, B, c_mod, C, reuse_a, reuse_b)
  return __builtin_amdgcn_wmma_f32_16x16x32_f16(false, a, false, b, (short)0, c,
                                                false, false);
}

// ---------------------------------------------------------------------------
// 1) fp32 -> f16 cast (weights), vectorized
// ---------------------------------------------------------------------------
__global__ void cast_f16_kernel(const float* __restrict__ src,
                                _Float16* __restrict__ dst, int n4) {
  int i = blockIdx.x * blockDim.x + threadIdx.x;
  if (i < n4) {
    float4 v = ((const float4*)src)[i];
    h4 o = { (_Float16)v.x, (_Float16)v.y, (_Float16)v.z, (_Float16)v.w };
    ((h4*)dst)[i] = o;
  }
}

// ---------------------------------------------------------------------------
// 2) LayerNorm (one 256-thread block per row of 4096) -> f16
// ---------------------------------------------------------------------------
__global__ __launch_bounds__(256)
void ln_f16_kernel(const float* __restrict__ x, const float* __restrict__ w,
                   _Float16* __restrict__ y) {
  __shared__ float red[2][8];
  const int s = blockIdx.x;
  const int tid = threadIdx.x;
  const float* row = x + (size_t)s * HID;

  float4 v[4];
  float sum = 0.f, sq = 0.f;
#pragma unroll
  for (int i = 0; i < 4; ++i) {
    v[i] = ((const float4*)row)[tid + 256 * i];
    sum += v[i].x + v[i].y + v[i].z + v[i].w;
    sq  += v[i].x * v[i].x + v[i].y * v[i].y + v[i].z * v[i].z + v[i].w * v[i].w;
  }
#pragma unroll
  for (int off = 16; off > 0; off >>= 1) {
    sum += __shfl_xor(sum, off, 32);
    sq  += __shfl_xor(sq,  off, 32);
  }
  if ((tid & 31) == 0) { red[0][tid >> 5] = sum; red[1][tid >> 5] = sq; }
  __syncthreads();
  float ts = 0.f, tq = 0.f;
#pragma unroll
  for (int i = 0; i < 8; ++i) { ts += red[0][i]; tq += red[1][i]; }
  const float mu  = ts * (1.0f / HID);
  const float var = tq * (1.0f / HID) - mu * mu;
  const float rs  = rsqrtf(var + 1e-5f);

#pragma unroll
  for (int i = 0; i < 4; ++i) {
    int idx = (tid + 256 * i) * 4;
    const float* wv = w + idx;
    h4 o = { (_Float16)((v[i].x - mu) * rs * wv[0]),
             (_Float16)((v[i].y - mu) * rs * wv[1]),
             (_Float16)((v[i].z - mu) * rs * wv[2]),
             (_Float16)((v[i].w - mu) * rs * wv[3]) };
    *(h4*)(y + (size_t)s * HID + idx) = o;
  }
}

// ---------------------------------------------------------------------------
// 3) NT GEMM with WMMA:  C[M,N] = A[M,K] * B[N,K]^T   (A,B f16 row-major)
//    Block tile 128x128, K-step 64, 8 waves: each wave 32(M) x 64(N).
// ---------------------------------------------------------------------------
template <typename OT>
__global__ __launch_bounds__(256)
void gemm_nt_kernel(const _Float16* __restrict__ A, const _Float16* __restrict__ B,
                    OT* __restrict__ C, int M, int N, int K) {
  __shared__ _Float16 As[128 * 72];   // 72-halfword stride: 144B rows (16B aligned)
  __shared__ _Float16 Bs[128 * 72];

  const int tid  = threadIdx.x;
  const int lane = tid & 31;
  const int wave = tid >> 5;
  const int wm   = (wave >> 1) * 32;   // 0..96
  const int wn   = (wave & 1) * 64;    // 0,64
  const int m0   = blockIdx.y * 128;
  const int n0   = blockIdx.x * 128;
  const int ln16 = lane & 15;
  const int hi8  = (lane >> 4) * 8;    // K-chunk select per half-wave
  const int mh   = (lane >> 4) * 8;    // C-layout row offset per half-wave

  v8f acc[2][4];
#pragma unroll
  for (int i = 0; i < 2; ++i)
#pragma unroll
    for (int j = 0; j < 4; ++j) acc[i][j] = vzero8();

  for (int k0 = 0; k0 < K; k0 += 64) {
#pragma unroll
    for (int it = 0; it < 4; ++it) {            // 1024 16B chunks / 256 threads
      int i = tid + it * 256;
      int r = i >> 3, c = (i & 7) * 8;
      *(uint4*)&As[r * 72 + c] = *(const uint4*)(A + (size_t)(m0 + r) * K + k0 + c);
      *(uint4*)&Bs[r * 72 + c] = *(const uint4*)(B + (size_t)(n0 + r) * K + k0 + c);
    }
    __syncthreads();

#pragma unroll
    for (int ks = 0; ks < 64; ks += 32) {
      v16h a2[2];
#pragma unroll
      for (int mt = 0; mt < 2; ++mt) {
        const _Float16* p = &As[(wm + mt * 16 + ln16) * 72 + ks + hi8];
        a2[mt] = frag_ld(p, p + 16);
      }
#pragma unroll
      for (int nt = 0; nt < 4; ++nt) {
        const _Float16* p = &Bs[(wn + nt * 16 + ln16) * 72 + ks + hi8];
        v16h b = frag_ld(p, p + 16);
        acc[0][nt] = wmma16(a2[0], b, acc[0][nt]);
        acc[1][nt] = wmma16(a2[1], b, acc[1][nt]);
      }
    }
    __syncthreads();
  }

#pragma unroll
  for (int mt = 0; mt < 2; ++mt)
#pragma unroll
    for (int nt = 0; nt < 4; ++nt) {
      int col = n0 + wn + nt * 16 + ln16;
#pragma unroll
      for (int r = 0; r < 8; ++r) {
        int row = m0 + wm + mt * 16 + mh + r;
        C[(size_t)row * N + col] = (OT)acc[mt][nt][r];
      }
    }
}

// ---------------------------------------------------------------------------
// 4) RoPE in place on Q heads and K head of each group (f16 storage)
//    qkv layout: [S][G=8][768] where group = 4*128 Q | 128 K | 128 V
// ---------------------------------------------------------------------------
__global__ void rope_kernel(_Float16* __restrict__ qkv,
                            const float* __restrict__ cosp,
                            const float* __restrict__ sinp) {
  const int s    = blockIdx.x;
  const int t    = blockIdx.y;             // 0..39 : 8 groups x (4 Q + 1 K)
  const int g    = t / 5, slot = t % 5;
  const int colb = (slot < 4) ? slot * HD : 512;
  _Float16* base = qkv + (size_t)s * OQKV + g * 768 + colb;
  const int j = threadIdx.x;               // 0..63
  float x1 = (float)base[j];
  float x2 = (float)base[j + 64];
  float c  = cosp[s * 64 + j];
  float sn = sinp[s * 64 + j];
  base[j]      = (_Float16)(x1 * c - x2 * sn);
  base[j + 64] = (_Float16)(x2 * c + x1 * sn);
}

// ---------------------------------------------------------------------------
// 5) Flash-style causal GQA attention.
//    grid = (S/128, 32 heads); block = 256 (8 waves, 16 q-rows each).
//    Online softmax kept in the native WMMA C layout:
//      lane L, VGPR r  ->  (m = r + 8*(L>=16), n = L&15)
//    so row statistics reduce across the 16-lane halves.
// ---------------------------------------------------------------------------
__global__ __launch_bounds__(256)
void attn_kernel(const _Float16* __restrict__ qkv, _Float16* __restrict__ ctx) {
  __shared__ _Float16 Kt[64 * 136];        // K tile  [64 keys][128 d], 272B rows
  __shared__ _Float16 Vt[128 * 72];        // V tile transposed [128 d][64 keys]
  __shared__ _Float16 Ps[8 * 16 * 72];     // per-wave P staging [16 rows][64 keys]

  const int tid  = threadIdx.x;
  const int lane = tid & 31;
  const int wave = tid >> 5;
  const int ln16 = lane & 15;
  const int hi8  = (lane >> 4) * 8;
  const int mh   = (lane >> 4) * 8;
  const int head = blockIdx.y;
  const int g    = head >> 2;
  const size_t qoff = (size_t)g * 768 + (size_t)(head & 3) * HD;
  const size_t koff = (size_t)g * 768 + 512;
  const size_t voff = (size_t)g * 768 + 640;
  const int qr0 = blockIdx.x * 128 + wave * 16;

  // Q fragments for this wave's 16 rows: 4 x (16x32) over d=128.
  v16h qa[4];
#pragma unroll
  for (int kd = 0; kd < 4; ++kd) {
    const _Float16* p = qkv + (size_t)(qr0 + ln16) * OQKV + qoff + kd * 32 + hi8;
    qa[kd] = frag_ld(p, p + 16);
  }

  v8f o[8];
#pragma unroll
  for (int i = 0; i < 8; ++i) o[i] = vzero8();
  float m[8], l[8];
#pragma unroll
  for (int r = 0; r < 8; ++r) { m[r] = -1e30f; l[r] = 0.f; }

  const int nkt = 2 * (blockIdx.x + 1);       // causal: tiles up to diagonal
  for (int kt = 0; kt < nkt; ++kt) {
    const int kbase = kt * 64;

    // Stage K tile (row-major) and V tile (transposed) into LDS.
#pragma unroll
    for (int it = 0; it < 4; ++it) {          // 1024 16B chunks / 256 threads
      int i = tid + it * 256;
      int r = i >> 4, c = (i & 15) * 8;
      *(uint4*)&Kt[r * 136 + c] =
          *(const uint4*)(qkv + (size_t)(kbase + r) * OQKV + koff + c);
      h8 vv = *(const h8*)(qkv + (size_t)(kbase + r) * OQKV + voff + c);
#pragma unroll
      for (int j = 0; j < 8; ++j) Vt[(c + j) * 72 + r] = vv[j];
    }
    __syncthreads();

    // S = Q * K^T : 4 n-tiles x 4 k-steps of WMMA.
    v8f st[4];
#pragma unroll
    for (int nt = 0; nt < 4; ++nt) st[nt] = vzero8();
#pragma unroll
    for (int kd = 0; kd < 4; ++kd)
#pragma unroll
      for (int nt = 0; nt < 4; ++nt) {
        const _Float16* p = &Kt[(nt * 16 + ln16) * 136 + kd * 32 + hi8];
        v16h b = frag_ld(p, p + 16);
        st[nt] = wmma16(qa[kd], b, st[nt]);
      }

    // Scale + causal mask.
    const float scl = 0.08838834764831845f;   // 1/sqrt(128)
    const bool need_mask = (kbase + 63) > qr0;
#pragma unroll
    for (int nt = 0; nt < 4; ++nt) {
      int key = kbase + nt * 16 + ln16;
#pragma unroll
      for (int r = 0; r < 8; ++r) {
        float v = st[nt][r] * scl;
        if (need_mask && key > qr0 + mh + r) v = -1e30f;
        st[nt][r] = v;
      }
    }

    // Online softmax update (row reductions across 16-lane halves).
#pragma unroll
    for (int r = 0; r < 8; ++r) {
      float mv = fmaxf(fmaxf(st[0][r], st[1][r]), fmaxf(st[2][r], st[3][r]));
      mv = fmaxf(mv, __shfl_xor(mv, 1, 16));
      mv = fmaxf(mv, __shfl_xor(mv, 2, 16));
      mv = fmaxf(mv, __shfl_xor(mv, 4, 16));
      mv = fmaxf(mv, __shfl_xor(mv, 8, 16));
      float mn = fmaxf(m[r], mv);
      float al = __expf(m[r] - mn);
      m[r] = mn;
      float ps = 0.f;
#pragma unroll
      for (int nt = 0; nt < 4; ++nt) {
        float p = __expf(st[nt][r] - mn);
        st[nt][r] = p;
        ps += p;
      }
      ps += __shfl_xor(ps, 1, 16);
      ps += __shfl_xor(ps, 2, 16);
      ps += __shfl_xor(ps, 4, 16);
      ps += __shfl_xor(ps, 8, 16);
      l[r] = l[r] * al + ps;
#pragma unroll
      for (int dt = 0; dt < 8; ++dt) o[dt][r] *= al;
    }

    // P: C-layout -> A-layout via per-wave LDS staging (same-wave DS is ordered).
    _Float16* pb = &Ps[wave * 16 * 72];
#pragma unroll
    for (int nt = 0; nt < 4; ++nt)
#pragma unroll
      for (int r = 0; r < 8; ++r)
        pb[(mh + r) * 72 + nt * 16 + ln16] = (_Float16)st[nt][r];

    // O += P * V : 2 k-steps x 8 d-tiles of WMMA (V pre-transposed in LDS).
#pragma unroll
    for (int kk = 0; kk < 2; ++kk) {
      const _Float16* pa = &pb[ln16 * 72 + kk * 32 + hi8];
      v16h a = frag_ld(pa, pa + 16);
#pragma unroll
      for (int dt = 0; dt < 8; ++dt) {
        const _Float16* pv = &Vt[(dt * 16 + ln16) * 72 + kk * 32 + hi8];
        v16h b = frag_ld(pv, pv + 16);
        o[dt] = wmma16(a, b, o[dt]);
      }
    }
    __syncthreads();
  }

  // Normalize and store context as f16 [S][32*128].
#pragma unroll
  for (int dt = 0; dt < 8; ++dt) {
    int col = head * HD + dt * 16 + ln16;
#pragma unroll
    for (int r = 0; r < 8; ++r) {
      float v = o[dt][r] / l[r];
      ctx[(size_t)(qr0 + mh + r) * (NQH * HD) + col] = (_Float16)v;
    }
  }
}

// ---------------------------------------------------------------------------
// Host: orchestrate the pipeline (all launches on `stream`; ws layout fixed).
// ---------------------------------------------------------------------------
extern "C" void kernel_launch(void* const* d_in, const int* in_sizes, int n_in,
                              void* d_out, int out_size, void* d_ws, size_t ws_size,
                              hipStream_t stream) {
  const float* hs    = (const float*)d_in[0];   // [2048,1,4096]
  const float* lnw   = (const float*)d_in[1];   // [4096]
  const float* wqkvf = (const float*)d_in[2];   // [6144,4096]
  const float* wprof = (const float*)d_in[3];   // [4096,4096]
  const float* cosp  = (const float*)d_in[4];   // [2048,64]
  const float* sinp  = (const float*)d_in[5];   // [2048,64]
  float* out = (float*)d_out;                   // [2048,4096] f32

  char* ws = (char*)d_ws;
  _Float16* xln  = (_Float16*)(ws);                       // 16 MB
  _Float16* wq16 = (_Float16*)(ws + ((size_t)16  << 20)); // 48 MB
  _Float16* wp16 = (_Float16*)(ws + ((size_t)64  << 20)); // 32 MB
  _Float16* qkv  = (_Float16*)(ws + ((size_t)96  << 20)); // 24 MB
  _Float16* ctx  = (_Float16*)(ws + ((size_t)120 << 20)); // 16 MB (ends 136 MB)

  // Weight casts (recomputed every call; deterministic).
  {
    int n4 = OQKV * HID / 4;
    cast_f16_kernel<<<(n4 + 255) / 256, 256, 0, stream>>>(wqkvf, wq16, n4);
  }
  {
    int n4 = HID * HID / 4;
    cast_f16_kernel<<<(n4 + 255) / 256, 256, 0, stream>>>(wprof, wp16, n4);
  }

  // LayerNorm -> f16 activations.
  ln_f16_kernel<<<S_LEN, 256, 0, stream>>>(hs, lnw, xln);

  // QKV projection: [2048,4096] x [6144,4096]^T -> f16 [2048,6144].
  gemm_nt_kernel<_Float16><<<dim3(OQKV / 128, S_LEN / 128), 256, 0, stream>>>(
      xln, wq16, qkv, S_LEN, OQKV, HID);

  // RoPE on Q and K heads.
  rope_kernel<<<dim3(S_LEN, NKVG * 5), 64, 0, stream>>>(qkv, cosp, sinp);

  // Causal GQA flash attention -> f16 context [2048,4096].
  attn_kernel<<<dim3(S_LEN / 128, NQH), 256, 0, stream>>>(qkv, ctx);

  // Output projection: [2048,4096] x [4096,4096]^T -> f32 output.
  gemm_nt_kernel<float><<<dim3(HID / 128, S_LEN / 128), 256, 0, stream>>>(
      ctx, wp16, out, S_LEN, HID, HID);
}
